// SpGraphAttentionLayer_88510686036767
// MI455X (gfx1250) — matrix operand order repaired and checked by reading, not compiled, not run on hardware
//
#include <hip/hip_runtime.h>
#include <hip/hip_bf16.h>

typedef __attribute__((ext_vector_type(16))) __bf16 v16bf;
typedef __attribute__((ext_vector_type(8)))  __bf16 v8bf;
typedef __attribute__((ext_vector_type(4)))  __bf16 v4bf;
typedef __attribute__((ext_vector_type(8)))  float  v8f;

#define N_NODES 100000
#define E_EDGES 3200000
#define FIN     256
#define FOUT    128
#define ALPHA   0.2f

// ---- helpers ----------------------------------------------------------------

// f32 -> bf16 round-to-nearest-even
static __device__ __forceinline__ __bf16 f2bf(float f) {
    unsigned u = __builtin_bit_cast(unsigned, f);
    unsigned r = (u + 0x7FFFu + ((u >> 16) & 1u)) >> 16;
    return __builtin_bit_cast(__bf16, (unsigned short)r);
}

// monotonic float<->uint mapping so float max == uint max (for atomicMax u32)
static __device__ __forceinline__ unsigned fmap(float f) {
    unsigned u = __builtin_bit_cast(unsigned, f);
    return (u & 0x80000000u) ? ~u : (u | 0x80000000u);
}
static __device__ __forceinline__ float funmap(unsigned u) {
    unsigned v = (u & 0x80000000u) ? (u & 0x7FFFFFFFu) : ~u;
    return __builtin_bit_cast(float, v);
}

// ---- kernel 0: out=0, mapped max=0 (== -inf), sums=0, W -> bf16 transposed --

__global__ void gat_init(float* __restrict__ out, unsigned* __restrict__ m_u,
                         float* __restrict__ ssum, const float* __restrict__ W,
                         __bf16* __restrict__ Wt) {
    int stride = gridDim.x * blockDim.x;
    for (int i = blockIdx.x * blockDim.x + threadIdx.x; i < N_NODES * FOUT; i += stride) {
        out[i] = 0.0f;
        if (i < N_NODES) { m_u[i] = 0u; ssum[i] = 0.0f; }
        if (i < FOUT * FIN) {
            int n = i >> 8, k = i & 255;          // Wt[n][k] = W[k][n]
            Wt[i] = f2bf(W[k * FOUT + n]);
        }
    }
}

// ---- kernel 1: h = x @ W via v_wmma_f32_16x16x32_bf16 -----------------------
// Block = 16 rows of x (LDS, padded stride for conflict-free b128 reads),
// 8 waves = 8 col tiles of 16. Per k-step each lane issues 2x ds_load_b128 (A)
// and 2x global_load_b128 from col-major Wt (B); k loop fully unrolled -> 8
// WMMAs pipelined against the loads.

__global__ __launch_bounds__(256) void gat_gemm_wmma(
    const float* __restrict__ x, const __bf16* __restrict__ Wt,
    float* __restrict__ h) {
    constexpr int LDSW = FIN + 8;                 // 264: 4-bank rotate per row
    __shared__ __bf16 xs[16 * LDSW];
    const int rowbase = blockIdx.x * 16;

    // stage x tile: float4 loads, packed v4bf stores
    const float4* x4 = (const float4*)(x + (size_t)rowbase * FIN);
    for (int i = threadIdx.x; i < (16 * FIN) / 4; i += 256) {
        float4 v = x4[i];
        int row = (i * 4) >> 8, col = (i * 4) & 255;
        v4bf p = {f2bf(v.x), f2bf(v.y), f2bf(v.z), f2bf(v.w)};
        *(v4bf*)&xs[row * LDSW + col] = p;
    }
    __syncthreads();

    const int lane  = threadIdx.x & 31;
    const int g     = lane >> 4;                  // lane-half select
    const int mr    = lane & 15;                  // A row / B,C col within tile
    const int nbase = (threadIdx.x >> 5) * 16;
    const __bf16* wrow = Wt + (size_t)(nbase + mr) * FIN;
    const __bf16* arow = xs + mr * LDSW;

    v8f c = {};
#pragma unroll
    for (int kb = 0; kb < FIN; kb += 32) {
        v8bf alo = *(const v8bf*)&arow[kb + g * 8];
        v8bf ahi = *(const v8bf*)&arow[kb + 16 + g * 8];
        v8bf blo = *(const v8bf*)&wrow[kb + g * 8];
        v8bf bhi = *(const v8bf*)&wrow[kb + 16 + g * 8];
        v16bf a = __builtin_shufflevector(alo, ahi,
                    0,1,2,3,4,5,6,7,8,9,10,11,12,13,14,15);
        v16bf b = __builtin_shufflevector(blo, bhi,
                    0,1,2,3,4,5,6,7,8,9,10,11,12,13,14,15);
        c = __builtin_amdgcn_wmma_f32_16x16x32_bf16(
                /*neg_a=*/false, a, /*neg_b=*/false, b,
                /*c_mod=*/(short)0, c, /*reuse_a=*/false, /*reuse_b=*/false);
    }
    // D: lane holds col nbase+mr, rows g*8 + r
#pragma unroll
    for (int r = 0; r < 8; ++r)
        h[(rowbase + g * 8 + r) * FOUT + nbase + mr] = c[r];
}

// ---- kernel 2: a_src[i] = h[i,:].attn[:128]; a_dst[i] = h[i,:].attn[128:] ---

__global__ __launch_bounds__(256) void gat_attn_dots(
    const float* __restrict__ h, const float* __restrict__ attn,
    float* __restrict__ a_src, float* __restrict__ a_dst) {
    const int lane = threadIdx.x & 31;
    const int wid  = blockIdx.x * (blockDim.x >> 5) + (threadIdx.x >> 5);
    const int nw   = gridDim.x * (blockDim.x >> 5);
    const float4* h4  = (const float4*)h;
    const float4* at4 = (const float4*)attn;
    const float4 as = at4[lane];        // attn[0..127]
    const float4 ad = at4[32 + lane];   // attn[128..255]
    for (int row = wid; row < N_NODES; row += nw) {
        float4 v = h4[row * 32 + lane];
        float ds = v.x * as.x + v.y * as.y + v.z * as.z + v.w * as.w;
        float dd = v.x * ad.x + v.y * ad.y + v.z * ad.z + v.w * ad.w;
#pragma unroll
        for (int off = 16; off > 0; off >>= 1) {
            ds += __shfl_xor(ds, off, 32);
            dd += __shfl_xor(dd, off, 32);
        }
        if (lane == 0) { a_src[row] = ds; a_dst[row] = dd; }
    }
}

// ---- kernel 3: edge scores + segment max (atomic u32 max on mapped floats) --

__global__ void gat_edge_score(const int* __restrict__ edge,
                               const float* __restrict__ a_src,
                               const float* __restrict__ a_dst,
                               float* __restrict__ score,
                               unsigned* __restrict__ m_u) {
    int stride = gridDim.x * blockDim.x;
    for (int e = blockIdx.x * blockDim.x + threadIdx.x; e < E_EDGES; e += stride) {
        int s = edge[e];
        int d = edge[E_EDGES + e];
        float v  = a_src[s] + a_dst[d];
        float sc = (v > 0.0f) ? -v : -(ALPHA * v);   // -leakyrelu
        score[e] = sc;
        atomicMax(&m_u[s], fmap(sc));
    }
}

// ---- kernel 4: e = exp(score - m[src]); segment sum via atomic f32 add ------

__global__ void gat_edge_exp(const int* __restrict__ edge,
                             const unsigned* __restrict__ m_u,
                             float* __restrict__ score_then_e,
                             float* __restrict__ ssum) {
    int stride = gridDim.x * blockDim.x;
    for (int e = blockIdx.x * blockDim.x + threadIdx.x; e < E_EDGES; e += stride) {
        int s = edge[e];
        float ex = __expf(score_then_e[e] - funmap(m_u[s]));
        score_then_e[e] = ex;
        atomicAdd(&ssum[s], ex);
    }
}

// ---- kernel 5: out[src,:] += (e/s[src]) * h[dst,:], one wave32 per edge -----
// h/out rows are L2-resident (51 MB each << 192 MB L2); atomics stay on-chip.
// coef computed once on lane 0 and broadcast.

__global__ __launch_bounds__(256) void gat_aggregate(
    const int* __restrict__ edge, const float* __restrict__ ebuf,
    const float* __restrict__ ssum, const float* __restrict__ h,
    float* __restrict__ out) {
    const int lane = threadIdx.x & 31;
    const int wid  = blockIdx.x * (blockDim.x >> 5) + (threadIdx.x >> 5);
    const int nw   = gridDim.x * (blockDim.x >> 5);
    const float4* h4 = (const float4*)h;
    for (int e = wid; e < E_EDGES; e += nw) {
        int s = edge[e];
        int d = edge[E_EDGES + e];
        float coef = 0.0f;
        if (lane == 0) coef = ebuf[e] / ssum[s];
        coef = __shfl(coef, 0, 32);
        float4 v = h4[d * 32 + lane];
        float* o = out + s * FOUT + lane * 4;
        atomicAdd(o + 0, coef * v.x);
        atomicAdd(o + 1, coef * v.y);
        atomicAdd(o + 2, coef * v.z);
        atomicAdd(o + 3, coef * v.w);
    }
}

// ---- kernel 6: in-place ELU -------------------------------------------------

__global__ void gat_elu(float* __restrict__ out) {
    int stride = gridDim.x * blockDim.x;
    for (int i = blockIdx.x * blockDim.x + threadIdx.x; i < N_NODES * FOUT; i += stride) {
        float v = out[i];
        out[i] = v > 0.0f ? v : (__expf(v) - 1.0f);
    }
}

// ---- launch -----------------------------------------------------------------

extern "C" void kernel_launch(void* const* d_in, const int* in_sizes, int n_in,
                              void* d_out, int out_size, void* d_ws, size_t ws_size,
                              hipStream_t stream) {
    const float* x    = (const float*)d_in[0];
    const int*   edge = (const int*)d_in[1];   // [2, E] row-major
    const float* W    = (const float*)d_in[2];
    const float* attn = (const float*)d_in[3];
    float* out = (float*)d_out;

    // workspace layout (all offsets 16B-aligned)
    char* ws = (char*)d_ws;
    float*    h     = (float*)(ws);                                  // 51.2 MB
    float*    ebuf  = (float*)(ws + (size_t)N_NODES * FOUT * 4);     // 12.8 MB
    float*    a_src = (float*)(ws + 64000000);                       // 0.4 MB
    float*    a_dst = (float*)(ws + 64400000);                       // 0.4 MB
    unsigned* m_u   = (unsigned*)(ws + 64800000);                    // 0.4 MB
    float*    ssum  = (float*)(ws + 65200000);                       // 0.4 MB
    __bf16*   Wt    = (__bf16*)(ws + 65600000);                      // 64 KB (col-major W)

    gat_init<<<50000, 256, 0, stream>>>(out, m_u, ssum, W, Wt);
    gat_gemm_wmma<<<N_NODES / 16, 256, 0, stream>>>(x, Wt, h);
    gat_attn_dots<<<12500, 256, 0, stream>>>(h, attn, a_src, a_dst);
    gat_edge_score<<<12500, 256, 0, stream>>>(edge, a_src, a_dst, ebuf, m_u);
    gat_edge_exp<<<12500, 256, 0, stream>>>(edge, m_u, ebuf, ssum);
    gat_aggregate<<<8192, 256, 0, stream>>>(edge, ebuf, ssum, h, out);
    gat_elu<<<50000, 256, 0, stream>>>(out);
}
